// VectorQuantizer_70763881168914
// MI455X (gfx1250) — matrix operand order repaired and checked by reading, not compiled, not run on hardware
//
#include <hip/hip_runtime.h>
#include <hip/hip_bf16.h>
#include <math.h>

typedef __attribute__((ext_vector_type(16))) _Float16 v16h;
typedef __attribute__((ext_vector_type(8)))  _Float16 v8h;
typedef __attribute__((ext_vector_type(8)))  float    v8f;

union V16 { v16h v; v8h h[2]; };

#if defined(__has_builtin)
#if __has_builtin(__builtin_amdgcn_sched_barrier)
#define SCHED_FENCE() __builtin_amdgcn_sched_barrier(0)
#endif
#endif
#ifndef SCHED_FENCE
#define SCHED_FENCE() asm volatile("" ::: "memory")
#endif

#define N_ROWS   65536     // 16*64*64
#define C_DIM    64
#define NE       2048
#define RB_ROWS  16        // rows per block-iteration (one WMMA M-tile)
#define NRB      4096      // N_ROWS / RB_ROWS
#define NWG      512       // persistent workgroups for kernel 2
#define LOG_NE   7.624618986159398f   // log(2048)
#define BETA     0.25f

// d_out layout (floats): zq[4194304] | commit | kl | lb | cb[131072] | perplexity
#define OFF_ZQ      0
#define OFF_COMMIT  4194304
#define OFF_KL      4194305
#define OFF_LB      4194306
#define OFF_CB      4194307
#define OFF_PERP    4325379

// workspace layout (bytes)
#define WS_CBH      0            // 2048*64 halves         = 262144 B
#define WS_PP       262144       // NWG*2048 f32           = 4194304 B
#define WS_PM       4456448      // NWG f32                = 2048 B
#define WS_CNT      4458496     // 2048 u32                = 8192 B
#define WS_BP       4466688      // 8*3 f32 block partials

// ---------------------------------------------------------------------------
// Kernel 1: normalize codebook rows. One wave (32 lanes) per row.
// Writes f32 cb (output) and f16 cbh (WMMA operand). Block 0 zeroes counts.
// ---------------------------------------------------------------------------
__global__ void vq_k1_codebook(const float* __restrict__ emb,
                               float* __restrict__ cb,
                               _Float16* __restrict__ cbh,
                               unsigned* __restrict__ counts) {
  const int tid  = threadIdx.x;
  const int lane = tid & 31;
  const int row  = blockIdx.x * 8 + (tid >> 5);

  float v0 = emb[row * 64 + lane];
  float v1 = emb[row * 64 + 32 + lane];
  float ss = v0 * v0 + v1 * v1;
  #pragma unroll
  for (int m = 16; m >= 1; m >>= 1) ss += __shfl_xor(ss, m, 32);
  const float inv = 1.0f / fmaxf(sqrtf(ss), 1e-12f);
  const float c0 = v0 * inv, c1 = v1 * inv;
  cb [row * 64 + lane]      = c0;
  cb [row * 64 + 32 + lane] = c1;
  cbh[row * 64 + lane]      = (_Float16)c0;
  cbh[row * 64 + 32 + lane] = (_Float16)c1;

  if (blockIdx.x == 0) {
    #pragma unroll
    for (int i = 0; i < 8; ++i) counts[tid + i * 256] = 0u;
  }
}

// ---------------------------------------------------------------------------
// Kernel 2: main WMMA kernel. 256 threads = 8 waves. Persistent over row
// blocks. Each wave owns 256 columns (16 tiles of 16); d tile-block kept in
// registers; argmax/softmax/column-prob-sums via wave32 shuffles + small LDS.
// sched_barrier every 4 tiles bounds B-fragment staging so the wave stays
// under 256 VGPRs (no s_set_vgpr_msb churn, better occupancy).
// ---------------------------------------------------------------------------
__global__ void __launch_bounds__(256)
vq_k2_main(const float* __restrict__ z,
           const float* __restrict__ cb_f32,
           const _Float16* __restrict__ cbh,
           float* __restrict__ zq,
           float* __restrict__ partial_p,
           float* __restrict__ partial_m,
           unsigned* __restrict__ counts) {
  __shared__ float    zs[RB_ROWS][64];
  __shared__ _Float16 znh[RB_ROWS][72];     // padded stride: LDS-bank friendly
  __shared__ float    invn[RB_ROWS];
  __shared__ float    lds_rm[RB_ROWS][8];
  __shared__ int      lds_rc[RB_ROWS][8];
  __shared__ float    lds_fm[RB_ROWS];
  __shared__ int      lds_idx[RB_ROWS];
  __shared__ float    lds_s[RB_ROWS][8];
  __shared__ float    lds_invS[RB_ROWS];
  __shared__ float    paccum[NE];

  const int tid  = threadIdx.x;
  const int w    = tid >> 5;       // wave id 0..7
  const int lane = tid & 31;
  const int half = lane >> 4;      // 0: lanes 0-15, 1: lanes 16-31
  const int l16  = lane & 15;

  for (int i = tid; i < NE; i += 256) paccum[i] = 0.0f;
  float msum = 0.0f;
  __syncthreads();

  for (int rb = blockIdx.x; rb < NRB; rb += NWG) {
    const int row_base = rb * RB_ROWS;

    // ---- stage 16 z rows into LDS (coalesced) ----
    #pragma unroll
    for (int i = 0; i < 4; ++i) {
      int e = tid + i * 256;
      zs[e >> 6][e & 63] = z[row_base * 64 + e];
    }
    __syncthreads();
    if (tid < RB_ROWS) {
      float ss = 0.0f;
      #pragma unroll
      for (int c = 0; c < 64; ++c) ss += zs[tid][c] * zs[tid][c];
      invn[tid] = 1.0f / fmaxf(sqrtf(ss), 1e-12f);
    }
    __syncthreads();
    #pragma unroll
    for (int i = 0; i < 4; ++i) {
      int e = tid + i * 256, r = e >> 6, c = e & 63;
      znh[r][c] = (_Float16)(zs[r][c] * invn[r]);
    }
    __syncthreads();

    // ---- A fragments (16-bit A 16x32 layout: lane m=l16, K-half by lane/16)
    const int m  = l16;
    const int hb = 8 * half;
    V16 A0, A1;
    A0.h[0] = *(const v8h*)&znh[m][hb];
    A0.h[1] = *(const v8h*)&znh[m][16 + hb];
    A1.h[0] = *(const v8h*)&znh[m][32 + hb];
    A1.h[1] = *(const v8h*)&znh[m][48 + hb];

    // ---- 16 column tiles x (2 WMMA, K=64) ----
    v8f acc[16];
    const _Float16* bbase = cbh + (w * 256 + l16) * 64 + half * 8;
    #pragma unroll
    for (int T = 0; T < 16; ++T) {
      const _Float16* bp = bbase + T * (16 * 64);
      V16 B0, B1;
      B0.h[0] = *(const v8h*)(bp);           // K 8h..8h+7, 16+8h..16+8h+7 per lane half
      B0.h[1] = *(const v8h*)(bp + 16);
      B1.h[0] = *(const v8h*)(bp + 32);      // K 32..63
      B1.h[1] = *(const v8h*)(bp + 48);
      v8f c = {};
      c = __builtin_amdgcn_wmma_f32_16x16x32_f16(false, A0.v, false, B0.v,
                                                 (short)0, c, false, false);
      c = __builtin_amdgcn_wmma_f32_16x16x32_f16(false, A1.v, false, B1.v,
                                                 (short)0, c, false, false);
      acc[T] = c;
      if ((T & 3) == 3) SCHED_FENCE();       // bound load-hoisting window
    }

    // ---- phase 1: row max + argmax (C layout: row = v + 8*half, col = l16)
    float bm[8]; int bcol[8];
    #pragma unroll
    for (int v = 0; v < 8; ++v) { bm[v] = acc[0][v]; bcol[v] = w * 256 + l16; }
    #pragma unroll
    for (int T = 1; T < 16; ++T) {
      const int cc = w * 256 + T * 16 + l16;
      #pragma unroll
      for (int v = 0; v < 8; ++v)
        if (acc[T][v] > bm[v]) { bm[v] = acc[T][v]; bcol[v] = cc; }
    }
    #pragma unroll
    for (int msk = 1; msk < 16; msk <<= 1) {
      #pragma unroll
      for (int v = 0; v < 8; ++v) {
        float ov = __shfl_xor(bm[v], msk, 32);
        int   oc = __shfl_xor(bcol[v], msk, 32);
        if (ov > bm[v] || (ov == bm[v] && oc < bcol[v])) { bm[v] = ov; bcol[v] = oc; }
      }
    }
    if (l16 == 0) {
      #pragma unroll
      for (int v = 0; v < 8; ++v) {
        lds_rm[v + 8 * half][w] = bm[v];
        lds_rc[v + 8 * half][w] = bcol[v];
      }
    }
    __syncthreads();
    if (tid < RB_ROWS) {
      float fm = lds_rm[tid][0]; int fc = lds_rc[tid][0];
      #pragma unroll
      for (int ww = 1; ww < 8; ++ww) {
        float ov = lds_rm[tid][ww]; int oc = lds_rc[tid][ww];
        if (ov > fm || (ov == fm && oc < fc)) { fm = ov; fc = oc; }
      }
      lds_fm[tid] = fm; lds_idx[tid] = fc;
    }
    __syncthreads();

    // ---- phase 2: softmax denominators (overwrite acc with exp values) ----
    float rm[8];
    #pragma unroll
    for (int v = 0; v < 8; ++v) rm[v] = lds_fm[v + 8 * half];
    float sv[8];
    #pragma unroll
    for (int v = 0; v < 8; ++v) sv[v] = 0.0f;
    #pragma unroll
    for (int T = 0; T < 16; ++T) {
      #pragma unroll
      for (int v = 0; v < 8; ++v) {
        float e = __expf(acc[T][v] - rm[v]);
        acc[T][v] = e;
        sv[v] += e;
      }
    }
    #pragma unroll
    for (int msk = 1; msk < 16; msk <<= 1) {
      #pragma unroll
      for (int v = 0; v < 8; ++v) sv[v] += __shfl_xor(sv[v], msk, 32);
    }
    if (l16 == 0) {
      #pragma unroll
      for (int v = 0; v < 8; ++v) lds_s[v + 8 * half][w] = sv[v];
    }
    __syncthreads();
    if (tid < RB_ROWS) {
      float S = 0.0f;
      #pragma unroll
      for (int ww = 0; ww < 8; ++ww) S += lds_s[tid][ww];
      lds_invS[tid] = 1.0f / S;
    }
    __syncthreads();

    // ---- phase 3: per-column prob sums over the 16 rows ----
    float is[8];
    #pragma unroll
    for (int v = 0; v < 8; ++v) is[v] = lds_invS[v + 8 * half];
    #pragma unroll
    for (int T = 0; T < 16; ++T) {
      float cc = 0.0f;
      #pragma unroll
      for (int v = 0; v < 8; ++v) cc += acc[T][v] * is[v];
      cc += __shfl_xor(cc, 16, 32);          // combine rows 0-7 and 8-15
      if (half == 0) paccum[w * 256 + T * 16 + l16] += cc;
    }

    // ---- z_q gather (f32 codebook), histogram, max-sum ----
    #pragma unroll
    for (int i = 0; i < 4; ++i) {
      int e = tid + i * 256, r = e >> 6, c = e & 63;
      zq[(row_base + r) * 64 + c] = cb_f32[lds_idx[r] * 64 + c];
    }
    if (tid < RB_ROWS) atomicAdd(&counts[lds_idx[tid]], 1u);
    if (tid == 0) {
      float s = 0.0f;
      #pragma unroll
      for (int r = 0; r < RB_ROWS; ++r) s += lds_fm[r];
      msum += s;
    }
    __syncthreads();
  }

  for (int i = tid; i < NE; i += 256)
    partial_p[(size_t)blockIdx.x * NE + i] = paccum[i];
  if (tid == 0) partial_m[blockIdx.x] = msum;
}

// ---------------------------------------------------------------------------
// Kernel 3: reduce partial_p over NWG -> p[j]; per-column loss terms,
// block-reduced; 8 blocks x 256 threads (one thread per codeword).
// ---------------------------------------------------------------------------
__global__ void vq_k3_cols(const float* __restrict__ partial_p,
                           const unsigned* __restrict__ counts,
                           float* __restrict__ blockpart) {
  __shared__ float sh[3][8];
  const int tid = threadIdx.x, lane = tid & 31, w = tid >> 5;
  const int j = blockIdx.x * 256 + tid;

  float s = 0.0f;
  for (int k = 0; k < NWG; ++k) s += partial_p[(size_t)k * NE + j];
  const float p = s / (float)N_ROWS;
  const float e = (float)counts[j] / (float)N_ROWS;

  float t1 = p * (logf(p) + LOG_NE);   // p*(log p - log(1/n_e))
  float t2 = e * p;                    // load-balancing term
  float t3 = e * logf(e + 1e-6f);      // entropy term for perplexity
  #pragma unroll
  for (int m = 16; m >= 1; m >>= 1) {
    t1 += __shfl_xor(t1, m, 32);
    t2 += __shfl_xor(t2, m, 32);
    t3 += __shfl_xor(t3, m, 32);
  }
  if (lane == 0) { sh[0][w] = t1; sh[1][w] = t2; sh[2][w] = t3; }
  __syncthreads();
  if (tid == 0) {
    float a = 0, b = 0, c = 0;
    #pragma unroll
    for (int ww = 0; ww < 8; ++ww) { a += sh[0][ww]; b += sh[1][ww]; c += sh[2][ww]; }
    blockpart[blockIdx.x * 3 + 0] = a;
    blockpart[blockIdx.x * 3 + 1] = b;
    blockpart[blockIdx.x * 3 + 2] = c;
  }
}

// ---------------------------------------------------------------------------
// Kernel 4: final scalars (single block, deterministic order).
// ---------------------------------------------------------------------------
__global__ void vq_k4_final(const float* __restrict__ partial_m,
                            const float* __restrict__ blockpart,
                            float* __restrict__ out) {
  __shared__ float sh[8];
  const int tid = threadIdx.x, lane = tid & 31, w = tid >> 5;
  float s = partial_m[tid] + partial_m[tid + 256];
  #pragma unroll
  for (int m = 16; m >= 1; m >>= 1) s += __shfl_xor(s, m, 32);
  if (lane == 0) sh[w] = s;
  __syncthreads();
  if (tid == 0) {
    float sum_m = 0.0f;
    #pragma unroll
    for (int ww = 0; ww < 8; ++ww) sum_m += sh[ww];
    float s1 = 0, s2 = 0, s3 = 0;
    for (int b = 0; b < 8; ++b) {
      s1 += blockpart[b * 3 + 0];
      s2 += blockpart[b * 3 + 1];
      s3 += blockpart[b * 3 + 2];
    }
    out[OFF_COMMIT] = (1.0f + BETA) * (1.0f - sum_m / (float)N_ROWS);
    out[OFF_KL]     = s1;
    out[OFF_LB]     = s2;
    out[OFF_PERP]   = expf(-s3);
  }
}

// ---------------------------------------------------------------------------
extern "C" void kernel_launch(void* const* d_in, const int* in_sizes, int n_in,
                              void* d_out, int out_size, void* d_ws, size_t ws_size,
                              hipStream_t stream) {
  const float* z   = (const float*)d_in[0];   // (16,64,64,64)
  const float* emb = (const float*)d_in[1];   // (2048,64)
  float* out = (float*)d_out;

  char* ws = (char*)d_ws;
  _Float16* cbh      = (_Float16*)(ws + WS_CBH);
  float*    pp       = (float*)(ws + WS_PP);
  float*    pm       = (float*)(ws + WS_PM);
  unsigned* counts   = (unsigned*)(ws + WS_CNT);
  float*    bp       = (float*)(ws + WS_BP);

  float* zq = out + OFF_ZQ;
  float* cb = out + OFF_CB;

  vq_k1_codebook<<<256, 256, 0, stream>>>(emb, cb, cbh, counts);
  vq_k2_main<<<NWG, 256, 0, stream>>>(z, cb, cbh, zq, pp, pm, counts);
  vq_k3_cols<<<8, 256, 0, stream>>>(pp, counts, bp);
  vq_k4_final<<<1, 256, 0, stream>>>(pm, bp, out);
}
